// NonLocalBlock3D_77111842833060
// MI455X (gfx1250) — matrix operand order repaired and checked by reading, not compiled, not run on hardware
//
#include <hip/hip_runtime.h>
#include <hip/hip_bf16.h>

typedef __attribute__((ext_vector_type(16))) __bf16 v16bf;
typedef __attribute__((ext_vector_type(8)))  __bf16 v8bf;
typedef __attribute__((ext_vector_type(8)))  float  v8f;
typedef __attribute__((ext_vector_type(4)))  int    v4i;

#define B_  8
#define C_  256
#define CI_ 128
#define T_  8
#define H_  28
#define W_  28
#define N_  (T_*H_*W_)     // 6272
#define HP_ 14
#define WP_ 14
#define NP_ (T_*HP_*WP_)   // 1568
#define NC_ (NP_/32)       // 49 key chunks

#if defined(__has_builtin)
#if __has_builtin(__builtin_amdgcn_global_load_async_to_lds_b128) && \
    __has_builtin(__builtin_amdgcn_s_wait_asynccnt)
#define NL_ASYNC 1
#endif
#endif

__device__ __forceinline__ void cp16_async(const __bf16* gsrc, __bf16* ldst) {
#ifdef NL_ASYNC
  typedef __attribute__((address_space(1))) v4i gvec_t;
  typedef __attribute__((address_space(3))) v4i lvec_t;
  __builtin_amdgcn_global_load_async_to_lds_b128(
      (gvec_t*)(v4i*)(__bf16*)gsrc, (lvec_t*)(v4i*)ldst, 0, 0);
#else
  *(v8bf*)ldst = *(const v8bf*)gsrc;  // fallback: sync load + ds_store
#endif
}

__device__ __forceinline__ void wait_async() {
#ifdef NL_ASYNC
  __builtin_amdgcn_s_wait_asynccnt(0);
#endif
}

__device__ __forceinline__ v8f wmma_bf16(v16bf a, v16bf b, v8f c) {
  return __builtin_amdgcn_wmma_f32_16x16x32_bf16(
      false, a, false, b, (short)0, c, false, false);
}

// A-fragment (16x32, bf16) from row-major bf16 [row][k]; ISA layout:
// lane<16: M=lane, V0-3=K(klo..klo+7), V4-7=K(klo+16..klo+23), klo=8*(lane>=16)
__device__ __forceinline__ v16bf load_a_bf(const __bf16* base, int rowstride,
                                           int kb, int lane) {
  int row = lane & 15;
  int klo = (lane >> 4) << 3;
  const __bf16* p = base + (size_t)row * rowstride + kb + klo;
  v8bf lo = *(const v8bf*)(p);
  v8bf hi = *(const v8bf*)(p + 16);
  v16bf a;
#pragma unroll
  for (int i = 0; i < 8; ++i) { a[i] = lo[i]; a[i + 8] = hi[i]; }
  return a;
}

// B-fragment (32x16, bf16) from "column-major-over-K" storage [col][k]:
// lane<16: N=lane, K=kb..kb+7,kb+16..kb+23; lane>=16: N=lane-16, K=kb+8..kb+15,kb+24..kb+31
__device__ __forceinline__ v16bf load_b_bf(const __bf16* base, int colstride,
                                           int kb, int lane) {
  int col = lane & 15;
  int k0  = kb + ((lane >> 4) << 3);
  const __bf16* p = base + (size_t)col * colstride + k0;
  v8bf lo = *(const v8bf*)(p);
  v8bf hi = *(const v8bf*)(p + 16);
  v16bf b;
#pragma unroll
  for (int i = 0; i < 8; ++i) { b[i] = lo[i]; b[i + 8] = hi[i]; }
  return b;
}

// ---------------------------------------------------------------------------
// Kernel 0: one-time fp32 -> bf16 conversion of all weight matrices so the
// GEMM kernels never burn VALU on v_cvt in their inner loops.
// ---------------------------------------------------------------------------
__global__ void nl_wcvt_kernel(const float* __restrict__ gw,
                               const float* __restrict__ tw,
                               const float* __restrict__ pw,
                               const float* __restrict__ Ww,
                               __bf16* __restrict__ gwb, __bf16* __restrict__ twb,
                               __bf16* __restrict__ pwb, __bf16* __restrict__ Wwb) {
  int i = blockIdx.x * 256 + threadIdx.x;
  if (i < CI_ * C_) {
    gwb[i] = (__bf16)gw[i];
    twb[i] = (__bf16)tw[i];
    pwb[i] = (__bf16)pw[i];
    Wwb[i] = (__bf16)Ww[i];  // C_*CI_ == CI_*C_
  }
}

// ---------------------------------------------------------------------------
// Kernel 1: the three 1x1x1 conv projections as WMMA GEMMs.
// Block = 256 threads (8 waves); one batch, one 16-wide spatial tile.
// Wave w owns M-tile w (16 ci rows). K = C = 256 (8 WMMA k-steps).
// ---------------------------------------------------------------------------
__global__ __launch_bounds__(256) void nl_proj_kernel(
    const float* __restrict__ x,
    const __bf16* __restrict__ gw, const float* __restrict__ gbias,
    const __bf16* __restrict__ tw, const float* __restrict__ tbias,
    const __bf16* __restrict__ pw, const float* __restrict__ pbias,
    __bf16* __restrict__ theta, __bf16* __restrict__ phiF,
    __bf16* __restrict__ gF) {
  constexpr int LDK = C_ + 8;  // padded K-stride (bf16) to spread LDS banks
  __shared__ __bf16 xs[16 * LDK];

  int blk   = blockIdx.x;
  int b     = blk / (N_ / 16);
  int nbase = (blk % (N_ / 16)) * 16;
  int tid   = threadIdx.x;

  // Cooperative load: x tile (256 channels x 16 positions) -> LDS bf16.
  {
    int n  = tid & 15;
    int c0 = tid >> 4;
    const float* xp = x + (size_t)b * C_ * N_ + nbase + n;
#pragma unroll
    for (int it = 0; it < 16; ++it) {
      int c = c0 + it * 16;
      xs[n * LDK + c] = (__bf16)xp[(size_t)c * N_];
    }
  }
  __syncthreads();

  int wave = tid >> 5, lane = tid & 31;
  int mrow = wave * 16;
  v8f cg = {}; v8f ct = {}; v8f cp = {};
#pragma unroll
  for (int ks = 0; ks < 8; ++ks) {
    int kb = ks * 32;
    v16bf bfrag = load_b_bf(xs, LDK, kb, lane);
    v16bf ag = load_a_bf(gw + (size_t)mrow * C_, C_, kb, lane);
    v16bf at = load_a_bf(tw + (size_t)mrow * C_, C_, kb, lane);
    v16bf ap = load_a_bf(pw + (size_t)mrow * C_, C_, kb, lane);
    cg = wmma_bf16(ag, bfrag, cg);
    ct = wmma_bf16(at, bfrag, ct);
    cp = wmma_bf16(ap, bfrag, cp);
  }

  int hi     = lane >> 4;
  int n      = nbase + (lane & 15);
  int cibase = mrow + hi * 8;
  size_t orow = ((size_t)b * N_ + n) * CI_ + cibase;
  v8bf vg, vt, vp;
#pragma unroll
  for (int j = 0; j < 8; ++j) {
    vg[j] = (__bf16)(cg[j] + gbias[cibase + j]);
    vt[j] = (__bf16)(ct[j] + tbias[cibase + j]);
    vp[j] = (__bf16)(cp[j] + pbias[cibase + j]);
  }
  *(v8bf*)(theta + orow) = vt;
  *(v8bf*)(phiF + orow)  = vp;
  *(v8bf*)(gF + orow)    = vg;
}

// ---------------------------------------------------------------------------
// Kernel 2: 2x2 spatial max-pool producing phiT (b,key,ci) and gT (b,ci,key).
// ---------------------------------------------------------------------------
__global__ void nl_pool_kernel(const __bf16* __restrict__ phiF,
                               const __bf16* __restrict__ gF,
                               __bf16* __restrict__ phiT,
                               __bf16* __restrict__ gT) {
  int idx = blockIdx.x * 256 + threadIdx.x;
  if (idx >= B_ * NP_ * CI_) return;
  int ci   = idx % CI_;
  int rest = idx / CI_;
  int key  = rest % NP_;
  int b    = rest / NP_;
  int t  = key / (HP_ * WP_);
  int hw = key % (HP_ * WP_);
  int h2 = hw / WP_, w2 = hw % WP_;
  float mp = -1e30f, mg = -1e30f;
#pragma unroll
  for (int dh = 0; dh < 2; ++dh)
#pragma unroll
    for (int dw = 0; dw < 2; ++dw) {
      int n = t * (H_ * W_) + (2 * h2 + dh) * W_ + (2 * w2 + dw);
      size_t off = ((size_t)b * N_ + n) * CI_ + ci;
      float vp = (float)phiF[off]; mp = vp > mp ? vp : mp;
      float vg = (float)gF[off];   mg = vg > mg ? vg : mg;
    }
  phiT[((size_t)b * NP_ + key) * CI_ + ci] = (__bf16)mp;
  gT[((size_t)b * CI_ + ci) * NP_ + key]   = (__bf16)mg;
}

// ---------------------------------------------------------------------------
// Kernel 3: fused flash attention with double-buffered async K/V staging.
// Each chunk (32 keys) of phiT/gT is pulled into LDS once per block via
// GLOBAL_LOAD_ASYNC_TO_LDS_B128 (ASYNCcnt) while the previous chunk computes;
// all 8 waves share it. S = Q*Phi (8 WMMAs), online softmax via 16-lane
// shuffles, P through per-wave LDS (C-layout -> A-layout), Y += P*G (8 WMMAs).
// ---------------------------------------------------------------------------
__global__ __launch_bounds__(256) void nl_attn_kernel(
    const __bf16* __restrict__ theta, const __bf16* __restrict__ phiT,
    const __bf16* __restrict__ gT, __bf16* __restrict__ y) {
  constexpr int PHI_LD = CI_ + 8;  // 136 bf16 per key row (padded)
  constexpr int G_LD   = 32 + 8;   // 40 bf16 per channel row (padded)
  __shared__ __bf16 sphi[2][32 * PHI_LD];
  __shared__ __bf16 sg[2][CI_ * G_LD];
  __shared__ __bf16 pbuf[8][16 * 32];

  int blk  = blockIdx.x;
  int b    = blk / (N_ / 128);
  int qt   = blk % (N_ / 128);
  int tid  = threadIdx.x, wave = tid >> 5, lane = tid & 31;
  int hi   = lane >> 4;
  int qbase = qt * 128 + wave * 16;

  const __bf16* th = theta + (size_t)b * N_ * CI_;
  const __bf16* ph = phiT + (size_t)b * NP_ * CI_;
  const __bf16* gv = gT + (size_t)b * CI_ * NP_;
  __bf16* pb = pbuf[wave];

  // Async staging of one 32-key chunk: 512 x b128 for phi, 512 x b128 for g.
  auto issue_chunk = [&](int kc, int buf) {
#pragma unroll
    for (int r = 0; r < 2; ++r) {
      int i = tid + r * 256;              // phi transfer id 0..511
      int key = i >> 4, seg = i & 15;     // 16 x 16B per key row
      cp16_async(ph + (size_t)(kc * 32 + key) * CI_ + seg * 8,
                 &sphi[buf][key * PHI_LD + seg * 8]);
    }
#pragma unroll
    for (int r = 0; r < 2; ++r) {
      int i = tid + r * 256;              // g transfer id 0..511
      int ch = i >> 2, seg = i & 3;       // 4 x 16B per channel row
      cp16_async(gv + (size_t)ch * NP_ + kc * 32 + seg * 8,
                 &sg[buf][ch * G_LD + seg * 8]);
    }
  };

  // Preload Q fragments (K = CI = 128 -> 4 k-steps of 32).
  v16bf qa[4];
#pragma unroll
  for (int ks = 0; ks < 4; ++ks)
    qa[ks] = load_a_bf(th + (size_t)qbase * CI_, CI_, ks * 32, lane);

  v8f acc[8];
#pragma unroll
  for (int nt = 0; nt < 8; ++nt) acc[nt] = (v8f){};
  float m[8], l[8];
#pragma unroll
  for (int j = 0; j < 8; ++j) { m[j] = -1e30f; l[j] = 0.0f; }

  issue_chunk(0, 0);
  for (int kc = 0; kc < NC_; ++kc) {
    wait_async();       // this wave's transfers for chunk kc are done
    __syncthreads();    // all waves' transfers visible; buf reuse is safe
    int buf = kc & 1;
    if (kc + 1 < NC_) issue_chunk(kc + 1, buf ^ 1);

    v8f s0 = {}, s1 = {};
#pragma unroll
    for (int ks = 0; ks < 4; ++ks) {
      v16bf b0 = load_b_bf(&sphi[buf][0], PHI_LD, ks * 32, lane);
      v16bf b1 = load_b_bf(&sphi[buf][16 * PHI_LD], PHI_LD, ks * 32, lane);
      s0 = wmma_bf16(qa[ks], b0, s0);
      s1 = wmma_bf16(qa[ks], b1, s1);
    }
    // Online softmax; row M = j + 8*hi lives across 16 lanes of each half.
#pragma unroll
    for (int j = 0; j < 8; ++j) {
      float mx = fmaxf(s0[j], s1[j]);
#pragma unroll
      for (int off = 1; off < 16; off <<= 1) mx = fmaxf(mx, __shfl_xor(mx, off, 32));
      float mnew  = fmaxf(m[j], mx);
      float p0    = __expf(s0[j] - mnew);
      float p1    = __expf(s1[j] - mnew);
      float rs    = p0 + p1;
#pragma unroll
      for (int off = 1; off < 16; off <<= 1) rs += __shfl_xor(rs, off, 32);
      float alpha = __expf(m[j] - mnew);
      l[j] = l[j] * alpha + rs;
      m[j] = mnew;
#pragma unroll
      for (int nt = 0; nt < 8; ++nt) acc[nt][j] *= alpha;
      int row = j + 8 * hi;
      pb[row * 32 + (lane & 15)]      = (__bf16)p0;
      pb[row * 32 + 16 + (lane & 15)] = (__bf16)p1;
    }
    v16bf pa = load_a_bf(pb, 32, 0, lane);
#pragma unroll
    for (int nt = 0; nt < 8; ++nt) {
      v16bf gb = load_b_bf(&sg[buf][(nt * 16) * G_LD], G_LD, 0, lane);
      acc[nt] = wmma_bf16(pa, gb, acc[nt]);
    }
  }

  __bf16* yb = y + (size_t)b * N_ * CI_;
#pragma unroll
  for (int nt = 0; nt < 8; ++nt) {
#pragma unroll
    for (int j = 0; j < 8; ++j) {
      int q = qbase + j + 8 * hi;
      yb[(size_t)q * CI_ + nt * 16 + (lane & 15)] = (__bf16)(acc[nt][j] / l[j]);
    }
  }
}

// ---------------------------------------------------------------------------
// Kernel 4: output projection W*y + b + residual x (fp32 out).
// ---------------------------------------------------------------------------
__global__ __launch_bounds__(256) void nl_out_kernel(
    const __bf16* __restrict__ Ww, const float* __restrict__ Wb,
    const __bf16* __restrict__ y, const float* __restrict__ x,
    float* __restrict__ out) {
  int blk   = blockIdx.x;
  int b     = blk / (N_ / 16);
  int nbase = (blk % (N_ / 16)) * 16;
  int tid = threadIdx.x, wave = tid >> 5, lane = tid & 31;
  int hi  = lane >> 4;
  const __bf16* yb = y + ((size_t)b * N_ + nbase) * CI_;
#pragma unroll
  for (int half = 0; half < 2; ++half) {
    int mrow = (wave + half * 8) * 16;
    v8f c = {};
#pragma unroll
    for (int ks = 0; ks < 4; ++ks) {
      v16bf a  = load_a_bf(Ww + (size_t)mrow * CI_, CI_, ks * 32, lane);
      v16bf bb = load_b_bf(yb, CI_, ks * 32, lane);
      c = wmma_bf16(a, bb, c);
    }
    int n      = nbase + (lane & 15);
    int cobase = mrow + hi * 8;
#pragma unroll
    for (int j = 0; j < 8; ++j) {
      int co = cobase + j;
      size_t off = ((size_t)b * C_ + co) * N_ + n;
      out[off] = c[j] + Wb[co] + x[off];
    }
  }
}

// ---------------------------------------------------------------------------
extern "C" void kernel_launch(void* const* d_in, const int* in_sizes, int n_in,
                              void* d_out, int out_size, void* d_ws,
                              size_t ws_size, hipStream_t stream) {
  const float* x   = (const float*)d_in[0];
  const float* gw  = (const float*)d_in[1];
  const float* gb  = (const float*)d_in[2];
  const float* tw  = (const float*)d_in[3];
  const float* tb  = (const float*)d_in[4];
  const float* pw  = (const float*)d_in[5];
  const float* pbv = (const float*)d_in[6];
  const float* Ww  = (const float*)d_in[7];
  const float* Wb  = (const float*)d_in[8];
  float* out = (float*)d_out;

  char* ws = (char*)d_ws;
  const size_t szFull = (size_t)B_ * N_ * CI_ * sizeof(__bf16);   // 12.8 MB
  const size_t szPool = (size_t)B_ * NP_ * CI_ * sizeof(__bf16);  // 3.2 MB
  const size_t szW    = (size_t)CI_ * C_ * sizeof(__bf16);        // 64 KB
  __bf16* theta = (__bf16*)ws; ws += szFull;
  __bf16* phiF  = (__bf16*)ws; ws += szFull;
  __bf16* gF    = (__bf16*)ws; ws += szFull;
  __bf16* phiT  = (__bf16*)ws; ws += szPool;
  __bf16* gT    = (__bf16*)ws; ws += szPool;
  __bf16* yws   = (__bf16*)ws; ws += szFull;
  __bf16* gwb   = (__bf16*)ws; ws += szW;
  __bf16* twb   = (__bf16*)ws; ws += szW;
  __bf16* pwb   = (__bf16*)ws; ws += szW;
  __bf16* Wwb   = (__bf16*)ws; ws += szW;

  nl_wcvt_kernel<<<(CI_ * C_ + 255) / 256, 256, 0, stream>>>(
      gw, tw, pw, Ww, gwb, twb, pwb, Wwb);
  nl_proj_kernel<<<B_ * (N_ / 16), 256, 0, stream>>>(
      x, gwb, gb, twb, tb, pwb, pbv, theta, phiF, gF);
  nl_pool_kernel<<<(B_ * NP_ * CI_ + 255) / 256, 256, 0, stream>>>(
      phiF, gF, phiT, gT);
  nl_attn_kernel<<<B_ * (N_ / 128), 256, 0, stream>>>(theta, phiT, gT, yws);
  nl_out_kernel<<<B_ * (N_ / 16), 256, 0, stream>>>(Wwb, Wb, yws, x, out);
}
